// DirectMPIGO_flow0_74732430950536
// MI455X (gfx1250) — compile-verified, matrix operand order checked by reading, not run on hardware
//
#include <hip/hip_runtime.h>
#include <hip/hip_bf16.h>
#include <math.h>

// ---------------- problem constants ----------------
#define RAYS   8192
#define DSAMP  128
#define NXg    160
#define NYg    160
#define NZg    128
#define GSZ    (NXg*NYg*NZg)       // per-channel grid stride
#define PDIMc  16
#define K0c    12
#define WID    128
#define RGBFEA 78
#define DIM0   39                  // 3 + 3*4*2 + 12
#define DIM0P  64                  // padded to 2 K-tiles of 32
#define VSRc   2.0f                // 256/128
#define LAMc   0.01f

typedef __attribute__((ext_vector_type(16))) _Float16 v16h;
typedef __attribute__((ext_vector_type(8)))  float    v8f;

// Load one 16x32 f16 WMMA fragment from a row-major LDS matrix.
// For A: "row" = M rows of the tile. For B (stored transposed, [N][K]): "row" = N.
// Layout per CDNA5 ISA 7.12.2 (16-bit A 16x32):
//   lanes 0-15 : row = tileBase+lane,  K = {0..7} in v0-3, {16..23} in v4-7
//   lanes 16-31: row = tileBase+lane-16, K = {8..15} in v0-3, {24..31} in v4-7
__device__ __forceinline__ v16h ldsFrag(const _Float16* buf, int ldHalfs,
                                        int tileBase, int kBase) {
    const int lane = threadIdx.x & 31;
    const int r    = tileBase + (lane & 15);
    const int hi   = (lane >> 4) * 8;          // K sub-offset 0 or 8
    const unsigned* p32 = (const unsigned*)buf;
    const int ldw  = ldHalfs >> 1;
    const int rowW = r * ldw;
    union { v16h v; unsigned u[8]; } f;
#pragma unroll
    for (int j = 0; j < 8; ++j) {
        const int k = (j < 4) ? (hi + 2*j) : (16 + hi + 2*(j - 4));
        f.u[j] = p32[rowW + ((kBase + k) >> 1)];
    }
    return f.v;
}

__device__ __forceinline__ float sigmoidf_(float x) {
    return 1.0f / (1.0f + __expf(-x));
}

__global__ void __launch_bounds__(256, 1)
nerf_fused_kernel(const float* __restrict__ rays_o, const float* __restrict__ rays_d,
                  const float* __restrict__ dgrid,  const float* __restrict__ pgrid,
                  const float* __restrict__ k0grid,
                  const float* __restrict__ gflow_den, const float* __restrict__ gflow_rgb,
                  const float* __restrict__ w0, const float* __restrict__ b0,
                  const float* __restrict__ w1, const float* __restrict__ b1,
                  const float* __restrict__ w2, const float* __restrict__ b2,
                  const float* __restrict__ act_shift,
                  float* __restrict__ out) {
    // ---------------- LDS ----------------
    __shared__ __align__(16) _Float16 Xlds[DSAMP * DIM0P];     // 16 KB  X  [128][64]
    __shared__ __align__(16) _Float16 W0T [WID  * DIM0P];      // 16 KB  w0^T [128][64] (K padded w/ 0)
    __shared__ __align__(16) _Float16 H0  [DSAMP * WID];       // 32 KB  [128][128]
    __shared__ __align__(16) _Float16 W1T [WID  * WID];        // 32 KB  w1^T [128][128]
    __shared__ float b0l[WID], b1l[WID];
    __shared__ float w2l[WID * 3];                             // w2[:, :3] as [n][c]
    __shared__ float featAcc[DSAMP * 3];                       // layer-2 accumulator
    __shared__ float sb[DSAMP];                                // scan buffer
    __shared__ float wgtl[DSAMP];
    __shared__ float red[3 * DSAMP];
    __shared__ float sAinv;

    const int tid  = threadIdx.x;
    const int ray  = blockIdx.x;
    const int lane = tid & 31;
    const int wv   = tid >> 5;                                 // wave id == M-tile

    // ---------- phase 0: stage weights (all 256 threads) ----------
    for (int i = tid; i < WID * DIM0P; i += 256) {             // W0T[n][k] = w0[k][n]
        const int n = i >> 6, k = i & 63;
        W0T[i] = (_Float16)((k < DIM0) ? w0[k * WID + n] : 0.0f);
    }
    for (int i = tid; i < WID * WID; i += 256) {               // W1T[n][k] = w1[k][n]
        const int n = i >> 7, k = i & 127;
        W1T[i] = (_Float16)w1[k * WID + n];
    }
    for (int i = tid; i < WID; i += 256) { b0l[i] = b0[i]; b1l[i] = b1[i]; }
    for (int i = tid; i < WID * 3; i += 256) {
        const int k = i / 3, c = i - 3 * k;
        w2l[i] = w2[k * RGBFEA + c];
    }
    for (int i = tid; i < DSAMP * 3; i += 256) featAcc[i] = 0.0f;

    // ---------- phase 1: per-sample featurize (threads 0..127) ----------
    float alpha = 0.0f;
    if (tid < DSAMP) {
        const int s = tid;
        const float ox = rays_o[ray*3+0], oy = rays_o[ray*3+1], oz = rays_o[ray*3+2];
        const float dx = rays_d[ray*3+0], dy = rays_d[ray*3+1], dz = rays_d[ray*3+2];
        const float tt = (float)s * (1.0f / 127.0f);

        float px = fminf(fmaxf(ox + dx * tt, 0.0f), 1.0f);
        float py = fminf(fmaxf(oy + dy * tt, 0.0f), 1.0f);
        float pz = fminf(fmaxf(oz + dz * tt, 0.0f), 1.0f);

        const float gx = px * (float)(NXg - 1);
        const float gy = py * (float)(NYg - 1);
        const float gz = pz * (float)(NZg - 1);
        int ix = (int)gx; ix = ix > NXg-2 ? NXg-2 : ix;
        int iy = (int)gy; iy = iy > NYg-2 ? NYg-2 : iy;
        int iz = (int)gz; iz = iz > NZg-2 ? NZg-2 : iz;
        const float fx = gx - (float)ix, fy = gy - (float)iy, fz = gz - (float)iz;

        const int base = (ix * NYg + iy) * NZg + iz;
        int   cidx[8];
        float cw[8];
#pragma unroll
        for (int j = 0; j < 8; ++j) {
            const int jx = (j >> 2) & 1, jy = (j >> 1) & 1, jz = j & 1;
            cidx[j] = base + jx * (NYg * NZg) + jy * NZg + jz;
            cw[j]   = (jx ? fx : 1.0f - fx) * (jy ? fy : 1.0f - fy) * (jz ? fz : 1.0f - fz);
        }

        // density (1 ch)
        float dens = 0.0f;
#pragma unroll
        for (int j = 0; j < 8; ++j) dens += dgrid[cidx[j]] * cw[j];

        // params (16 ch)
        float prm[PDIMc];
#pragma unroll
        for (int c = 0; c < PDIMc; ++c) {
            const float* g = pgrid + c * GSZ;
            float v = 0.0f;
#pragma unroll
            for (int j = 0; j < 8; ++j) v += g[cidx[j]] * cw[j];
            prm[c] = v;
        }

        // k0 (12 ch)
        float k0f[K0c];
#pragma unroll
        for (int c = 0; c < K0c; ++c) {
            const float* g = k0grid + c * GSZ;
            float v = 0.0f;
#pragma unroll
            for (int j = 0; j < 8; ++j) v += g[cidx[j]] * cw[j];
            k0f[c] = v;
        }

        // activation chain
        float z = dens + gflow_den[0] * LAMc;
#pragma unroll
        for (int i = 0; i < 4; ++i) {
            const float p0 = prm[4*i+0], p1 = prm[4*i+1], p2 = prm[4*i+2], p3 = prm[4*i+3];
            z += p0 * tanhf(p1 * z + p2) * sigmoidf_(p3);
        }
        const float den = z + act_shift[s];
        const float sp  = (den > 20.0f) ? den : log1pf(__expf(den));
        alpha = 1.0f - __expf(-sp * VSRc);

        // view-dir embedding (same for all samples of this ray; cheap to recompute)
        const float nrm = sqrtf(dx*dx + dy*dy + dz*dz) + 1e-8f;
        const float vd[3] = { dx / nrm, dy / nrm, dz / nrm };
        _Float16* xr = &Xlds[s * DIM0P];
        xr[0] = (_Float16)vd[0]; xr[1] = (_Float16)vd[1]; xr[2] = (_Float16)vd[2];
#pragma unroll
        for (int c = 0; c < 3; ++c) {
#pragma unroll
            for (int i = 0; i < 4; ++i) {
                const float a = vd[c] * (float)(1 << i);
                xr[3  + c*4 + i] = (_Float16)__sinf(a);
                xr[15 + c*4 + i] = (_Float16)__cosf(a);
            }
        }
#pragma unroll
        for (int c = 0; c < K0c; ++c) xr[27 + c] = (_Float16)k0f[c];
#pragma unroll
        for (int c = DIM0; c < DIM0P; ++c) xr[c] = (_Float16)0.0f;
    }

    // ---------- phase 2: cumulative product scan over samples ----------
    float x = (tid < DSAMP) ? (1.0f - alpha + 1e-10f) : 1.0f;
    if (tid < DSAMP) sb[tid] = x;
    __syncthreads();
#pragma unroll
    for (int off = 1; off < DSAMP; off <<= 1) {
        float y = 1.0f;
        if (tid < DSAMP && tid >= off) y = sb[tid - off];
        __syncthreads();
        if (tid < DSAMP) { x *= y; sb[tid] = x; }
        __syncthreads();
    }
    if (tid < DSAMP) {
        const float texcl = (tid == 0) ? 1.0f : sb[tid - 1];
        wgtl[tid] = alpha * texcl;
        if (tid == DSAMP - 1) sAinv = x;     // T[:, -1]
    }
    __syncthreads();

    // ---------- phase 3: layer 0 GEMM  H0 = relu(X @ w0 + b0) ----------
    {
#pragma unroll
        for (int n = 0; n < 8; ++n) {
            v8f acc = {};
#pragma unroll
            for (int kt = 0; kt < 2; ++kt) {
                const v16h a = ldsFrag(Xlds, DIM0P, wv * 16, kt * 32);
                const v16h b = ldsFrag(W0T,  DIM0P, n  * 16, kt * 32);
                acc = __builtin_amdgcn_wmma_f32_16x16x32_f16(
                        false, a, false, b, (short)0, acc, false, false);
            }
            const int col = n * 16 + (lane & 15);
            const int hi8 = (lane >> 4) * 8;
            const float bias = b0l[col];
#pragma unroll
            for (int r = 0; r < 8; ++r) {
                const int row = wv * 16 + r + hi8;
                float h = acc[r] + bias;
                h = h > 0.0f ? h : 0.0f;
                H0[row * WID + col] = (_Float16)h;
            }
        }
    }
    __syncthreads();

    // ---------- phase 4: layer 1 GEMM fused with layer 2 (w2[:, :3]) ----------
    {
#pragma unroll
        for (int n = 0; n < 8; ++n) {
            v8f acc = {};
#pragma unroll
            for (int kt = 0; kt < 4; ++kt) {
                const v16h a = ldsFrag(H0,  WID, wv * 16, kt * 32);
                const v16h b = ldsFrag(W1T, WID, n  * 16, kt * 32);
                acc = __builtin_amdgcn_wmma_f32_16x16x32_f16(
                        false, a, false, b, (short)0, acc, false, false);
            }
            const int col = n * 16 + (lane & 15);
            const int hi8 = (lane >> 4) * 8;
            const float bias = b1l[col];
            const float wc0 = w2l[col * 3 + 0];
            const float wc1 = w2l[col * 3 + 1];
            const float wc2 = w2l[col * 3 + 2];
#pragma unroll
            for (int r = 0; r < 8; ++r) {
                const int row = wv * 16 + r + hi8;
                float h = acc[r] + bias;
                h = h > 0.0f ? h : 0.0f;
                atomicAdd(&featAcc[row * 3 + 0], h * wc0);   // ds_add_f32
                atomicAdd(&featAcc[row * 3 + 1], h * wc1);
                atomicAdd(&featAcc[row * 3 + 2], h * wc2);
            }
        }
    }
    __syncthreads();

    // ---------- phase 5: sigmoid, weight, per-ray reduction ----------
    if (tid < DSAMP) {
        const float w = wgtl[tid];
#pragma unroll
        for (int c = 0; c < 3; ++c) {
            const float f = featAcc[tid * 3 + c] + b2[c] + gflow_rgb[c] * LAMc;
            red[c * DSAMP + tid] = w * sigmoidf_(f);
        }
    }
    __syncthreads();
#pragma unroll
    for (int off = 64; off > 0; off >>= 1) {
        if (tid < off) {
            red[tid]             += red[tid + off];
            red[DSAMP + tid]     += red[DSAMP + tid + off];
            red[2 * DSAMP + tid] += red[2 * DSAMP + tid + off];
        }
        __syncthreads();
    }
    if (tid == 0) {
        const float ainv = sAinv;
        out[ray * 3 + 0] = red[0]         + ainv;
        out[ray * 3 + 1] = red[DSAMP]     + ainv;
        out[ray * 3 + 2] = red[2 * DSAMP] + ainv;
    }
}

extern "C" void kernel_launch(void* const* d_in, const int* in_sizes, int n_in,
                              void* d_out, int out_size, void* d_ws, size_t ws_size,
                              hipStream_t stream) {
    (void)in_sizes; (void)n_in; (void)out_size; (void)d_ws; (void)ws_size;
    const float* rays_o    = (const float*)d_in[0];
    const float* rays_d    = (const float*)d_in[1];
    const float* dgrid     = (const float*)d_in[2];
    const float* pgrid     = (const float*)d_in[3];
    const float* k0grid    = (const float*)d_in[4];
    const float* gflow_den = (const float*)d_in[5];
    const float* gflow_rgb = (const float*)d_in[6];
    const float* w0        = (const float*)d_in[7];
    const float* b0        = (const float*)d_in[8];
    const float* w1        = (const float*)d_in[9];
    const float* b1        = (const float*)d_in[10];
    const float* w2        = (const float*)d_in[11];
    const float* b2        = (const float*)d_in[12];
    const float* act_shift = (const float*)d_in[13];

    nerf_fused_kernel<<<RAYS, 256, 0, stream>>>(
        rays_o, rays_d, dgrid, pgrid, k0grid, gflow_den, gflow_rgb,
        w0, b0, w1, b1, w2, b2, act_shift, (float*)d_out);
}
